// AttackHead_10746008174759
// MI455X (gfx1250) — matrix operand order repaired and checked by reading, not compiled, not run on hardware
//
#include <hip/hip_runtime.h>
#include <hip/hip_bf16.h>
#include <stdint.h>

typedef __attribute__((ext_vector_type(16))) _Float16 v16h;
typedef __attribute__((ext_vector_type(8)))  float    v8f;
typedef __attribute__((ext_vector_type(4)))  unsigned int u32x4;
typedef __attribute__((ext_vector_type(8)))  int      i32x8;
typedef __attribute__((ext_vector_type(4)))  int      i32x4;

#define B_   8
#define N_   50000
#define E_   100000
#define D_   128      // embed dim
#define H_   128      // hidden dim
// First-layer GEMM K = 2*D_ = 256

// ---------------------------------------------------------------------------
// Weight prep: ew1/aw1 (f32 [256,128] row-major KxN) -> f16 WMMA B-fragment
// order for v_wmma_f32_16x16x32_f16.  B(32x16): lane = hi*16 + (N%16),
// this code's contiguous k order (A packed alike): pair j holds K = kt*32 + hi*16 + {2j,2j+1}.
// pair index r = ((kt*8 + nt)*32 + lane)*8 + j   (2 f16 per pair)
// ---------------------------------------------------------------------------
__global__ __launch_bounds__(256) void prep_weights(const float* __restrict__ ew1,
                                                    const float* __restrict__ aw1,
                                                    _Float16* __restrict__ wfrag) {
  int p = blockIdx.x * 256 + threadIdx.x;   // 0 .. 32767
  if (p >= 2 * 16384) return;
  int s    = p >> 14;
  int r    = p & 16383;
  int ktnt = r >> 8;
  int lane = (r >> 3) & 31;
  int j    = r & 7;
  int kt   = ktnt >> 3;
  int nt   = ktnt & 7;
  int col  = nt * 16 + (lane & 15);
  int hi   = lane >> 4;
  int K    = kt * 32 + hi * 16 + 2 * j;
  const float* w = s ? aw1 : ew1;
  size_t base = (size_t)s * 32768 + (size_t)r * 2;
  wfrag[base]     = (_Float16)w[K * H_ + col];
  wfrag[base + 1] = (_Float16)w[(K + 1) * H_ + col];
}

// ---------------------------------------------------------------------------
// Embedding prep: f32 node table -> f16 node table (fits in 192MB L2)
// ---------------------------------------------------------------------------
__global__ __launch_bounds__(256) void prep_embed(const float* __restrict__ src,
                                                  _Float16* __restrict__ dst) {
  long i = (long)blockIdx.x * 256 + threadIdx.x;   // one thread per 8 elements
  if (i * 8 >= (long)B_ * N_ * D_) return;
  const float4* s4 = (const float4*)src;
  float4 a = s4[i * 2];
  float4 b = s4[i * 2 + 1];
  union { _Float16 h[8]; uint4 u; } cv;
  cv.h[0] = (_Float16)a.x; cv.h[1] = (_Float16)a.y;
  cv.h[2] = (_Float16)a.z; cv.h[3] = (_Float16)a.w;
  cv.h[4] = (_Float16)b.x; cv.h[5] = (_Float16)b.y;
  cv.h[6] = (_Float16)b.z; cv.h[7] = (_Float16)b.w;
  ((uint4*)dst)[i] = cv.u;
}

// A-matrix (16x32 f16) pair index for element (row M, global k):
//   kt=k/32, kk=k%32; hi=kk>>4; j=(kk&15)>>1
//   pair index = (kt*32 + (M + hi*16))*8 + j
__device__ __forceinline__ int a_pair_idx(int k, int M) {
  int kt = k >> 5;
  int kk = k & 31;
  int hi = kk >> 4;
  int j  = (kk & 15) >> 1;
  return (kt * 32 + (M + hi * 16)) * 8 + j;
}

__device__ __forceinline__ void store_pairs(uint32_t* lds32, int k, int M, float4 v) {
  union { _Float16 h[4]; uint2 u; } cv;
  cv.h[0] = (_Float16)v.x; cv.h[1] = (_Float16)v.y;
  cv.h[2] = (_Float16)v.z; cv.h[3] = (_Float16)v.w;
  int p0 = a_pair_idx(k, M);              // k%4==0 -> even pair, 8B aligned
  *(uint2*)(lds32 + p0) = cv.u;
}

// ---------------------------------------------------------------------------
// TDM: DMA one contiguous 64KB block (16384 dwords) global -> LDS.
// D# built per CDNA5 ISA 8.3/8.4: count=1, type=2, data_size=4B,
// tile_dim0=tensor_dim0=tensor_dim0_stride=16384, tile_dim1=tensor_dim1=1.
// clang-23 toolchain form: 6 args (g0, g1, g2, g3, g4, cpol).
// ---------------------------------------------------------------------------
__device__ __forceinline__ void tdm_load_64k(const void* gsrc, uint32_t lds_byte_off) {
  unsigned long long ga = (unsigned long long)(uintptr_t)gsrc;
  u32x4 g0;
  g0[0] = 1u;                                            // count=1, user mode
  g0[1] = lds_byte_off;                                  // lds_addr [63:32]
  g0[2] = (unsigned int)ga;                              // global_addr lo
  g0[3] = ((unsigned int)(ga >> 32) & 0x01FFFFFFu)       // global_addr [56:32]
          | 0x80000000u;                                 // type=2 @ [127:126]
  i32x8 g1;
  g1[0] = 0x00020000;        // wg_mask=0, data_size=2 (4 bytes)
  g1[1] = 0x40000000;        // tensor_dim0 = 16384  (bits 79:48)
  g1[2] = 0x00010000;        // tensor_dim1 = 1      (bits 111:80)
  g1[3] = 0x40000000;        // tile_dim0   = 16384  (bits 127:112)
  g1[4] = 1;                 // tile_dim1 = 1, tile_dim2 = 0
  g1[5] = 16384;             // tensor_dim0_stride (bits 207:160)
  g1[6] = 0;
  g1[7] = 0;
  i32x4 z4 = {0, 0, 0, 0};
  i32x8 z8 = {0, 0, 0, 0, 0, 0, 0, 0};
  __builtin_amdgcn_tensor_load_to_lds(g0, g1, z4, z4, z8, 0);
}

// ---------------------------------------------------------------------------
// Fused gather + 2x(GEMM[16x256 x 256x128] + ReLU + head) per 16 edges.
// Block = 256 threads (8 wave32); wave w owns edges [ei0 + w*16, +16).
// Dynamic LDS: [0,64K) A-fragments, [64K,128K) staged B-fragments, then pen.
// ---------------------------------------------------------------------------
template <bool ASYNC_GATHER>
__global__ __launch_bounds__(256) void attack_head_kernel(
    const float*      __restrict__ emb,     // [B,N,128] f32 (fallback gather)
    const _Float16*   __restrict__ embH,    // [B,N,128] f16 (async gather)
    const long long*  __restrict__ edges,   // [B,E,2] i64
    const _Float16*   __restrict__ wfrag,   // [2][32768] f16 B-fragments (global)
    const float* __restrict__ eb1, const float* __restrict__ ew2, const float* __restrict__ eb2,
    const float* __restrict__ ab1, const float* __restrict__ aw2, const float* __restrict__ ab2,
    float* __restrict__ out_edge,           // [B,E]
    float* __restrict__ out_army) {         // [B,E,4]
  extern __shared__ char smem[];
  _Float16* afrag = (_Float16*)smem;               // 64KB: 8 waves * 16x256 f16
  _Float16* wbuf  = (_Float16*)(smem + 65536);     // 64KB: staged B-fragments
  float*    pen   = (float*)(smem + 131072);       // 512B

  const int tid  = threadIdx.x;
  const int wid  = tid >> 5;
  const int lane = tid & 31;
  const int b    = blockIdx.y;
  const int ei0  = blockIdx.x * 128 + wid * 16;

  // kick off TDM stage of edge-scorer weights ASAP (wave 0 only; one per block)
  if (wid == 0)
    tdm_load_64k(wfrag, (uint32_t)(uintptr_t)wbuf);

  // ---- per-edge metadata (each half-wave handles the same 16 edges) ----
  const int eLane = lane & 15;
  int myE = ei0 + eLane;
  int eC  = myE < E_ ? myE : (E_ - 1);
  const long long* ep = edges + (((size_t)b * E_ + (size_t)eC) << 1);
  long long s64 = ep[0];
  long long t64 = ep[1];
  const long long nmax = N_ - 1;
  int sc = (int)(s64 < 0 ? 0 : (s64 > nmax ? nmax : s64));
  int tc = (int)(t64 < 0 ? 0 : (t64 > nmax ? nmax : t64));
  if (lane < 16)
    pen[wid * 16 + eLane] = ((s64 == t64) && (s64 >= 0) && (t64 >= 0)) ? -100.0f : 0.0f;

  // ---- gather X (16 edges x 256 K) into A-fragment layout in LDS ----
  if (ASYNC_GATHER) {
    // lanes 0-15 DMA the src row of edge t, lanes 16-31 the tgt row; each lane
    // moves one 8-f16 K-group which is 16B-contiguous in A-fragment order.
    const _Float16* embRow = embH + (size_t)b * N_ * D_;
    uint32_t waveBase = (uint32_t)(uintptr_t)afrag + (uint32_t)(wid << 13);
    #pragma unroll
    for (int t = 0; t < 16; ++t) {
      int ns  = __shfl(sc, t, 32);
      int ntg = __shfl(tc, t, 32);
      int node = (lane < 16) ? ns : ntg;
      int grp  = lane & 15;
      int k    = ((lane >> 4) << 7) + (grp << 3);          // 0..127 or 128..255
      uint32_t ldsoff = waveBase + (uint32_t)(a_pair_idx(k, t) << 2);
      uint32_t goff   = (uint32_t)node * 256u + (uint32_t)(grp << 4);  // bytes
      asm volatile("global_load_async_to_lds_b128 %0, %1, %2"
                   :: "v"(ldsoff), "v"(goff), "s"(embRow) : "memory");
    }
    asm volatile("s_wait_asynccnt 0x0" ::: "memory");
  } else {
    uint32_t* lds32 = (uint32_t*)afrag + wid * 2048;
    #pragma unroll 2
    for (int e = 0; e < 16; ++e) {
      int sci = __shfl(sc, e, 32);
      int tci = __shfl(tc, e, 32);
      const float4* rs = (const float4*)(emb + ((size_t)b * N_ + (size_t)sci) * D_);
      const float4* rt = (const float4*)(emb + ((size_t)b * N_ + (size_t)tci) * D_);
      float4 vs = rs[lane];
      float4 vt = rt[lane];
      store_pairs(lds32, lane * 4,       e, vs);
      store_pairs(lds32, 128 + lane * 4, e, vt);
    }
  }

  if (wid == 0) __builtin_amdgcn_s_wait_tensorcnt(0);
  __syncthreads();

  // ---- load the 8 A-fragments for this wave's 16x256 tile ----
  v16h a[8];
  const v16h* av = (const v16h*)(afrag + wid * 4096);
  #pragma unroll
  for (int kt = 0; kt < 8; ++kt) a[kt] = av[kt * 32 + lane];

  const int col = lane & 15;
  const int hiM = lane >> 4;
  const v16h* wB = (const v16h*)wbuf;

  // ================= edge scorer: relu(X@ew1+eb1) @ ew2 =================
  float partE[8];
  #pragma unroll
  for (int r = 0; r < 8; ++r) partE[r] = 0.0f;

  #pragma unroll 1
  for (int nt = 0; nt < 8; ++nt) {
    v8f acc = {};
    #pragma unroll
    for (int kt = 0; kt < 8; ++kt) {
      v16h bf = wB[(kt * 8 + nt) * 32 + lane];   // ds_load_b128 x2
      acc = __builtin_amdgcn_wmma_f32_16x16x32_f16(
          false, a[kt], false, bf, (short)0, acc, false, false);
    }
    float bias = eb1[nt * 16 + col];
    float w2   = ew2[nt * 16 + col];
    #pragma unroll
    for (int r = 0; r < 8; ++r) {
      float h = acc[r] + bias;
      h = h > 0.0f ? h : 0.0f;
      partE[r] += h * w2;
    }
  }

  // restage: army-scorer weights over the same LDS buffer
  __syncthreads();
  if (wid == 0) {
    tdm_load_64k(wfrag + 32768, (uint32_t)(uintptr_t)wbuf);
    __builtin_amdgcn_s_wait_tensorcnt(0);
  }
  __syncthreads();

  // ================= army scorer: relu(X@aw1+ab1) @ aw2 =================
  float partA[8][4];
  #pragma unroll
  for (int r = 0; r < 8; ++r) {
    partA[r][0] = 0.0f; partA[r][1] = 0.0f;
    partA[r][2] = 0.0f; partA[r][3] = 0.0f;
  }

  const float4* aw2v = (const float4*)aw2;
  #pragma unroll 1
  for (int nt = 0; nt < 8; ++nt) {
    v8f acc = {};
    #pragma unroll
    for (int kt = 0; kt < 8; ++kt) {
      v16h bf = wB[(kt * 8 + nt) * 32 + lane];
      acc = __builtin_amdgcn_wmma_f32_16x16x32_f16(
          false, a[kt], false, bf, (short)0, acc, false, false);
    }
    float bias = ab1[nt * 16 + col];
    float4 w4  = aw2v[nt * 16 + col];
    #pragma unroll
    for (int r = 0; r < 8; ++r) {
      float h = acc[r] + bias;
      h = h > 0.0f ? h : 0.0f;
      partA[r][0] += h * w4.x;
      partA[r][1] += h * w4.y;
      partA[r][2] += h * w4.z;
      partA[r][3] += h * w4.w;
    }
  }

  // ---- reduce over 16 columns (xor<16 keeps the two row-halves apart) ----
  #pragma unroll
  for (int m = 1; m < 16; m <<= 1) {
    #pragma unroll
    for (int r = 0; r < 8; ++r) {
      partE[r]    += __shfl_xor(partE[r],    m, 32);
      partA[r][0] += __shfl_xor(partA[r][0], m, 32);
      partA[r][1] += __shfl_xor(partA[r][1], m, 32);
      partA[r][2] += __shfl_xor(partA[r][2], m, 32);
      partA[r][3] += __shfl_xor(partA[r][3], m, 32);
    }
  }

  // ---- writeback: lane 0 -> rows 0-7, lane 16 -> rows 8-15 ----
  if (col == 0) {
    float base2 = eb2[0];
    float c0 = ab2[0], c1 = ab2[1], c2 = ab2[2], c3 = ab2[3];
    #pragma unroll
    for (int r = 0; r < 8; ++r) {
      int M  = hiM * 8 + r;
      int ei = ei0 + M;
      if (ei < E_) {
        size_t gid = (size_t)b * E_ + (size_t)ei;
        out_edge[gid] = partE[r] + base2 + pen[wid * 16 + M];
        float4 o;
        o.x = partA[r][0] + c0;
        o.y = partA[r][1] + c1;
        o.z = partA[r][2] + c2;
        o.w = partA[r][3] + c3;
        *(float4*)(out_army + (gid << 2)) = o;
      }
    }
  }
}

extern "C" void kernel_launch(void* const* d_in, const int* in_sizes, int n_in,
                              void* d_out, int out_size, void* d_ws, size_t ws_size,
                              hipStream_t stream) {
  const float*     emb   = (const float*)d_in[0];
  const long long* edges = (const long long*)d_in[1];
  const float* ew1 = (const float*)d_in[2];
  const float* eb1 = (const float*)d_in[3];
  const float* ew2 = (const float*)d_in[4];
  const float* eb2 = (const float*)d_in[5];
  const float* aw1 = (const float*)d_in[6];
  const float* ab1 = (const float*)d_in[7];
  const float* aw2 = (const float*)d_in[8];
  const float* ab2 = (const float*)d_in[9];

  _Float16* wfrag = (_Float16*)d_ws;                       // 128KB fragments
  const size_t embh_off  = 131072;
  const size_t embh_size = sizeof(_Float16) * (size_t)B_ * N_ * D_;  // ~102MB
  _Float16* embH = (_Float16*)((char*)d_ws + embh_off);
  bool big = ws_size >= embh_off + embh_size;              // fixed per harness

  prep_weights<<<128, 256, 0, stream>>>(ew1, aw1, wfrag);

  float* out_edge = (float*)d_out;                         // [B,E]
  float* out_army = (float*)d_out + (size_t)B_ * E_;       // [B,E,4]

  dim3 grid((E_ + 127) / 128, B_);
  size_t shbytes = 131584;                                 // 64K + 64K + 512

  if (big) {
    long n8 = (long)B_ * N_ * D_ / 8;                      // 6.4M threads
    int blocks = (int)((n8 + 255) / 256);
    prep_embed<<<blocks, 256, 0, stream>>>(emb, embH);
    attack_head_kernel<true><<<grid, 256, shbytes, stream>>>(
        emb, embH, edges, wfrag, eb1, ew2, eb2, ab1, aw2, ab2, out_edge, out_army);
  } else {
    attack_head_kernel<false><<<grid, 256, shbytes, stream>>>(
        emb, embH, edges, wfrag, eb1, ew2, eb2, ab1, aw2, ab2, out_edge, out_army);
  }
}
